// GCNTransformer_34857954574425
// MI455X (gfx1250) — compile-verified, matrix operand order checked by reading, not run on hardware
//
#include <hip/hip_runtime.h>

// ---------------------------------------------------------------------------
// Types for CDNA5 WMMA (gfx1250, wave32)
// ---------------------------------------------------------------------------
typedef __attribute__((ext_vector_type(16))) __bf16 v16bf;
typedef __attribute__((ext_vector_type(8)))  float  v8f;

#define NEG_SLOPE 0.01f
#define BN_EPS    1e-5f

// ---------------------------------------------------------------------------
// Utility kernels
// ---------------------------------------------------------------------------
__global__ void k_zero(float* __restrict__ p, size_t n) {
  size_t i = (size_t)blockIdx.x * blockDim.x + threadIdx.x;
  size_t stride = (size_t)gridDim.x * blockDim.x;
  for (; i < n; i += stride) p[i] = 0.f;
}

__global__ void k_degree(const int* __restrict__ src, const float* __restrict__ ea,
                         float* __restrict__ deg, int E) {
  int i = blockIdx.x * blockDim.x + threadIdx.x;
  if (i < E) atomicAdd(&deg[src[i]], ea[i]);
}

__global__ void k_dis(const float* __restrict__ deg, float* __restrict__ dis, int N) {
  int i = blockIdx.x * blockDim.x + threadIdx.x;
  if (i < N) {
    float d = deg[i];
    dis[i] = d > 0.f ? rsqrtf(d) : 0.f;
  }
}

__global__ void k_wnorm(const int* __restrict__ src, const int* __restrict__ dst,
                        const float* __restrict__ ea, const float* __restrict__ dis,
                        float* __restrict__ w, int E) {
  int i = blockIdx.x * blockDim.x + threadIdx.x;
  if (i < E) w[i] = -dis[src[i]] * ea[i] * dis[dst[i]];
}

// out[dst, :] += w * v[src, :]   (4 channels per thread)
__global__ void k_spmv(const int* __restrict__ src, const int* __restrict__ dst,
                       const float* __restrict__ w, const float* __restrict__ v,
                       float* __restrict__ out, int E, int C) {
  size_t t = (size_t)blockIdx.x * blockDim.x + threadIdx.x;
  int per = C >> 2;
  size_t total = (size_t)E * per;
  if (t >= total) return;
  int e = (int)(t / per);
  int c = (int)(t % per) << 2;
  float wv = w[e];
  const float4 xi = *(const float4*)(v + (size_t)src[e] * C + c);
  float* o = out + (size_t)dst[e] * C + c;
  atomicAdd(o + 0, wv * xi.x);
  atomicAdd(o + 1, wv * xi.y);
  atomicAdd(o + 2, wv * xi.z);
  atomicAdd(o + 3, wv * xi.w);
}

// s[i] = 2*s[i] - x0[i]   (Chebyshev recurrence Tx2 = 2*L*Tx1 - Tx0)
__global__ void k_cheb2(float* __restrict__ s, const float* __restrict__ x0, size_t n) {
  size_t i = (size_t)blockIdx.x * blockDim.x + threadIdx.x;
  size_t stride = (size_t)gridDim.x * blockDim.x;
  for (; i < n; i += stride) s[i] = 2.f * s[i] - x0[i];
}

// ---------------------------------------------------------------------------
// Pack a fp32 weight matrix B[K,N] (row-major) into the CDNA5 WMMA B-fragment
// layout in bf16, zero-padded to a multiple of FOUR 16-column tiles (so the
// 16x64-per-wave GEMM can load B fragments unguarded for ragged N):
//   P[(((nt*KT + kt)*32 + lane)*16 + j]
//     lane: col = nt*16 + (lane&15); kh = (lane>>4)*8
//     j<8 : k = kt*32 + kh + j      j>=8 : k = kt*32 + 16 + kh + (j-8)
// Each lane's 16 bf16 (32 bytes) are contiguous -> 2 vector loads in the GEMM.
// ---------------------------------------------------------------------------
__global__ void k_pack_b(const float* __restrict__ B, __bf16* __restrict__ P,
                         int K, int N) {
  const int KT  = K >> 5;
  const int NT4 = (((N + 15) >> 4) + 3) & ~3;   // n-tiles padded to multiple of 4
  size_t total = (size_t)KT * NT4 * 512;
  size_t t = (size_t)blockIdx.x * blockDim.x + threadIdx.x;
  if (t >= total) return;
  int j    = (int)(t & 15);
  int lane = (int)((t >> 4) & 31);
  size_t tile = t >> 9;               // nt*KT + kt
  int kt = (int)(tile % KT);
  int nt = (int)(tile / KT);
  int col = (nt << 4) + (lane & 15);
  int kh  = (lane >> 4) << 3;
  int k   = (kt << 5) + kh + ((j < 8) ? j : j + 8);
  P[t] = (col < N) ? (__bf16)B[(size_t)k * N + col] : (__bf16)0.f;
}

// ---------------------------------------------------------------------------
// BatchNorm over rows (per-column mean/var), optional fused LeakyReLU
// one block (256 threads) per channel
// ---------------------------------------------------------------------------
template <bool LRELU>
__global__ __launch_bounds__(256) void k_bn(float* __restrict__ h,
                                            const float* __restrict__ g,
                                            const float* __restrict__ be,
                                            int M, int C) {
  const int c = blockIdx.x;
  const int tid = threadIdx.x;
  __shared__ float ssum[256];
  __shared__ float ssq[256];
  float s = 0.f, q = 0.f;
  for (int r = tid; r < M; r += 256) {
    float v = h[(size_t)r * C + c];
    s += v;
    q += v * v;
  }
  ssum[tid] = s;
  ssq[tid] = q;
  __syncthreads();
  for (int off = 128; off > 0; off >>= 1) {
    if (tid < off) {
      ssum[tid] += ssum[tid + off];
      ssq[tid]  += ssq[tid + off];
    }
    __syncthreads();
  }
  __shared__ float s_scale, s_shift;
  if (tid == 0) {
    float mu  = ssum[0] / (float)M;
    float var = ssq[0] / (float)M - mu * mu;
    float inv = rsqrtf(var + BN_EPS);
    float sc  = inv * g[c];
    s_scale = sc;
    s_shift = be[c] - mu * sc;
  }
  __syncthreads();
  float scale = s_scale, shift = s_shift;
  for (int r = tid; r < M; r += 256) {
    size_t idx = (size_t)r * C + c;
    float v = h[idx] * scale + shift;
    if (LRELU) v = v > 0.f ? v : NEG_SLOPE * v;
    h[idx] = v;
  }
}

// ---------------------------------------------------------------------------
// WMMA GEMM:  C[M,N] (+)= A[M,K] * B[K,N], bf16 A/B fragments, fp32 accum.
// Each wave computes a 16x64 output tile (4 x 16x16 accumulators), so one
// A fragment (loaded+converted once per K-step) feeds FOUR back-to-back
// v_wmma_f32_16x16x32_bf16 ops; WMMAs 2-4 set the reuse-A operand-cache
// hint (identical instruction, same A registers -> legal per ISA 7.12).
// 8 waves (256 threads) per block cover 128 rows x 64 cols.
// A: fp32 row-major, converted in-register; B: pre-packed bf16 fragments
// (padded to 4 n-tiles, so B loads need no guards even for N=10).
// ACC:  accumulate into existing C (3-term Cheb sum; always full-width N)
// RELU: fused ReLU epilogue
// ---------------------------------------------------------------------------
template <bool ACC, bool RELU>
__global__ __launch_bounds__(256) void k_wmma_gemm(const float* __restrict__ A,
                                                   const __bf16* __restrict__ Bp,
                                                   const float* __restrict__ bias,
                                                   float* __restrict__ C,
                                                   int M, int K, int N) {
  const int lane = threadIdx.x & 31;
  const int wave = threadIdx.x >> 5;
  const int m0  = (blockIdx.x * 8 + wave) * 16;
  const int nt0 = blockIdx.y * 4;          // first of 4 n-tiles for this wave
  if (m0 >= M) return;  // wave-uniform: EXEC stays all-ones for WMMA

  const int row = m0 + (lane & 15);
  const int kh  = (lane >> 4) << 3;        // 0 for lanes 0-15, 8 for lanes 16-31
  const int KT  = K >> 5;

  v8f acc0, acc1, acc2, acc3;
  if (ACC) {
#pragma unroll
    for (int i = 0; i < 8; ++i) {
      int r = m0 + ((lane < 16) ? i : (8 + i));
      const float* cr = C + (size_t)r * N + (nt0 << 4) + (lane & 15);
      acc0[i] = cr[0];
      acc1[i] = cr[16];
      acc2[i] = cr[32];
      acc3[i] = cr[48];
    }
  } else {
    float bv0 = 0.f, bv1 = 0.f, bv2 = 0.f, bv3 = 0.f;
    if (bias != nullptr) {
      int c0 = (nt0 << 4) + (lane & 15);
      bv0 = (c0      < N) ? bias[c0]      : 0.f;
      bv1 = (c0 + 16 < N) ? bias[c0 + 16] : 0.f;
      bv2 = (c0 + 32 < N) ? bias[c0 + 32] : 0.f;
      bv3 = (c0 + 48 < N) ? bias[c0 + 48] : 0.f;
    }
#pragma unroll
    for (int i = 0; i < 8; ++i) {
      acc0[i] = bv0; acc1[i] = bv1; acc2[i] = bv2; acc3[i] = bv3;
    }
  }

  const float* ap = A + (size_t)row * K + kh;
  const v16bf* bp0 = (const v16bf*)(Bp + (((size_t)(nt0 + 0) * KT) * 32 + lane) * 16);
  const v16bf* bp1 = (const v16bf*)(Bp + (((size_t)(nt0 + 1) * KT) * 32 + lane) * 16);
  const v16bf* bp2 = (const v16bf*)(Bp + (((size_t)(nt0 + 2) * KT) * 32 + lane) * 16);
  const v16bf* bp3 = (const v16bf*)(Bp + (((size_t)(nt0 + 3) * KT) * 32 + lane) * 16);

  for (int kt = 0; kt < KT; ++kt) {
    const float* a = ap + kt * 32;
    if (kt + 1 < KT) __builtin_prefetch(a + 32, 0, 2);

    const float4 a0 = *(const float4*)(a + 0);
    const float4 a1 = *(const float4*)(a + 4);
    const float4 a2 = *(const float4*)(a + 16);
    const float4 a3 = *(const float4*)(a + 20);
    const v16bf b0 = bp0[(size_t)kt * 32];
    const v16bf b1 = bp1[(size_t)kt * 32];
    const v16bf b2 = bp2[(size_t)kt * 32];
    const v16bf b3 = bp3[(size_t)kt * 32];

    v16bf af;
    af[0]  = (__bf16)a0.x;  af[1]  = (__bf16)a0.y;
    af[2]  = (__bf16)a0.z;  af[3]  = (__bf16)a0.w;
    af[4]  = (__bf16)a1.x;  af[5]  = (__bf16)a1.y;
    af[6]  = (__bf16)a1.z;  af[7]  = (__bf16)a1.w;
    af[8]  = (__bf16)a2.x;  af[9]  = (__bf16)a2.y;
    af[10] = (__bf16)a2.z;  af[11] = (__bf16)a2.w;
    af[12] = (__bf16)a3.x;  af[13] = (__bf16)a3.y;
    af[14] = (__bf16)a3.z;  af[15] = (__bf16)a3.w;

    acc0 = __builtin_amdgcn_wmma_f32_16x16x32_bf16(false, af, false, b0,
                                                   (short)0, acc0, false, false);
    acc1 = __builtin_amdgcn_wmma_f32_16x16x32_bf16(false, af, false, b1,
                                                   (short)0, acc1, true, false);
    acc2 = __builtin_amdgcn_wmma_f32_16x16x32_bf16(false, af, false, b2,
                                                   (short)0, acc2, true, false);
    acc3 = __builtin_amdgcn_wmma_f32_16x16x32_bf16(false, af, false, b3,
                                                   (short)0, acc3, true, false);
  }

  const int c0 = (nt0 << 4) + (lane & 15);
#pragma unroll
  for (int i = 0; i < 8; ++i) {
    int r = m0 + ((lane < 16) ? i : (8 + i));
    float* cr = C + (size_t)r * N + c0;
    float v0 = acc0[i], v1 = acc1[i], v2 = acc2[i], v3 = acc3[i];
    if (RELU) {
      v0 = v0 > 0.f ? v0 : 0.f;
      v1 = v1 > 0.f ? v1 : 0.f;
      v2 = v2 > 0.f ? v2 : 0.f;
      v3 = v3 > 0.f ? v3 : 0.f;
    }
    if (c0      < N) cr[0]  = v0;
    if (c0 + 16 < N) cr[16] = v1;
    if (c0 + 32 < N) cr[32] = v2;
    if (c0 + 48 < N) cr[48] = v3;
  }
}

// ---------------------------------------------------------------------------
// Host-side orchestration
// ---------------------------------------------------------------------------
static inline size_t pack_elems(int K, int N) {
  int NT4 = (((N + 15) >> 4) + 3) & ~3;
  return (size_t)(K >> 5) * (size_t)NT4 * 512;
}

extern "C" void kernel_launch(void* const* d_in, const int* in_sizes, int n_in,
                              void* d_out, int out_size, void* d_ws, size_t ws_size,
                              hipStream_t stream) {
  const float* x   = (const float*)d_in[0];
  const int*   ei  = (const int*)d_in[1];
  const float* ea  = (const float*)d_in[2];
  const float* W1  = (const float*)d_in[3];
  const float* b1  = (const float*)d_in[4];
  const float* g1  = (const float*)d_in[5];
  const float* be1 = (const float*)d_in[6];
  const float* W2  = (const float*)d_in[7];
  const float* b2  = (const float*)d_in[8];
  const float* g2  = (const float*)d_in[9];
  const float* be2 = (const float*)d_in[10];
  const float* Wf1 = (const float*)d_in[11];
  const float* bf1 = (const float*)d_in[12];
  const float* gf  = (const float*)d_in[13];
  const float* bef = (const float*)d_in[14];
  const float* Wf2 = (const float*)d_in[15];
  const float* bf2 = (const float*)d_in[16];
  float* out = (float*)d_out;

  const int CIN = 128, HID = 256, FC = 128, NC = 10;
  const int NV = in_sizes[0] / CIN;   // 50000
  const int E  = in_sizes[2];         // 800000
  const int* src = ei;
  const int* dst = ei + E;

  float* ws   = (float*)d_ws;
  float* deg  = ws;
  float* dis  = deg + NV;
  float* wnm  = dis + NV;
  float* bufA = wnm + E;                      // NV x 256
  float* bufB = bufA + (size_t)NV * HID;      // NV x 256
  float* bufH = bufB + (size_t)NV * HID;      // NV x 256
  float* bufC = bufH + (size_t)NV * HID;      // NV x 256

  const size_t pw1 = pack_elems(CIN, HID);    // per W1 slice
  const size_t pw2 = pack_elems(HID, HID);    // per W2 slice
  const size_t pf1 = pack_elems(HID, FC);
  __bf16* pk   = (__bf16*)(bufC + (size_t)NV * HID);
  __bf16* pW1  = pk;               // 3 slices
  __bf16* pW2  = pW1 + 3 * pw1;    // 3 slices
  __bf16* pWf1 = pW2 + 3 * pw2;
  __bf16* pWf2 = pWf1 + pf1;

  auto zero = [&](float* p, size_t n) {
    k_zero<<<2048, 256, 0, stream>>>(p, n);
  };
  auto pack = [&](const float* B, __bf16* P, int K, int N) {
    size_t tot = pack_elems(K, N);
    k_pack_b<<<(unsigned)((tot + 255) / 256), 256, 0, stream>>>(B, P, K, N);
  };
  auto spmv = [&](const float* vin, float* vout, int C) {
    zero(vout, (size_t)NV * C);
    size_t tot = (size_t)E * (size_t)(C >> 2);
    k_spmv<<<(unsigned)((tot + 255) / 256), 256, 0, stream>>>(src, dst, wnm, vin,
                                                              vout, E, C);
  };
  auto gemm = [&](const float* A, const __bf16* Bp, const float* bias, float* Cp,
                  int M, int K, int N, bool acc, bool relu) {
    dim3 gr((unsigned)(((M + 15) / 16 + 7) / 8), (unsigned)((N + 63) / 64));
    if (acc)
      k_wmma_gemm<true, false><<<gr, 256, 0, stream>>>(A, Bp, bias, Cp, M, K, N);
    else if (relu)
      k_wmma_gemm<false, true><<<gr, 256, 0, stream>>>(A, Bp, bias, Cp, M, K, N);
    else
      k_wmma_gemm<false, false><<<gr, 256, 0, stream>>>(A, Bp, bias, Cp, M, K, N);
  };

  // -------- pack all weights into WMMA B-fragment layout (bf16) --------
  pack(W1 + 0 * (size_t)CIN * HID, pW1 + 0 * pw1, CIN, HID);
  pack(W1 + 1 * (size_t)CIN * HID, pW1 + 1 * pw1, CIN, HID);
  pack(W1 + 2 * (size_t)CIN * HID, pW1 + 2 * pw1, CIN, HID);
  pack(W2 + 0 * (size_t)HID * HID, pW2 + 0 * pw2, HID, HID);
  pack(W2 + 1 * (size_t)HID * HID, pW2 + 1 * pw2, HID, HID);
  pack(W2 + 2 * (size_t)HID * HID, pW2 + 2 * pw2, HID, HID);
  pack(Wf1, pWf1, HID, FC);
  pack(Wf2, pWf2, FC, NC);

  // -------- edge normalization: w = -d^-1/2 * a * d^-1/2 --------
  zero(deg, (size_t)NV);
  k_degree<<<(E + 255) / 256, 256, 0, stream>>>(src, ea, deg, E);
  k_dis<<<(NV + 255) / 256, 256, 0, stream>>>(deg, dis, NV);
  k_wnorm<<<(E + 255) / 256, 256, 0, stream>>>(src, dst, ea, dis, wnm, E);

  // -------- ChebConv layer 1 (C_IN=128 -> HID=256) --------
  spmv(x, bufA, CIN);                                   // Tx1 = L*x
  spmv(bufA, bufB, CIN);                                // L*Tx1
  k_cheb2<<<2048, 256, 0, stream>>>(bufB, x, (size_t)NV * CIN);  // Tx2
  gemm(x,    pW1 + 0 * pw1, b1,      bufH, NV, CIN, HID, false, false);
  gemm(bufA, pW1 + 1 * pw1, nullptr, bufH, NV, CIN, HID, true,  false);
  gemm(bufB, pW1 + 2 * pw1, nullptr, bufH, NV, CIN, HID, true,  false);
  k_bn<true><<<HID, 256, 0, stream>>>(bufH, g1, be1, NV, HID);

  // -------- ChebConv layer 2 (HID -> HID) --------
  spmv(bufH, bufB, HID);                                // Tx1
  spmv(bufB, bufA, HID);                                // L*Tx1
  k_cheb2<<<2048, 256, 0, stream>>>(bufA, bufH, (size_t)NV * HID);  // Tx2
  gemm(bufH, pW2 + 0 * pw2, b2,      bufC, NV, HID, HID, false, false);
  gemm(bufB, pW2 + 1 * pw2, nullptr, bufC, NV, HID, HID, true,  false);
  gemm(bufA, pW2 + 2 * pw2, nullptr, bufC, NV, HID, HID, true,  false);
  k_bn<true><<<HID, 256, 0, stream>>>(bufC, g2, be2, NV, HID);

  // -------- MLP head --------
  gemm(bufC, pWf1, bf1, bufA, NV, HID, FC, false, true);   // FC1 + ReLU
  k_bn<false><<<FC, 256, 0, stream>>>(bufA, gf, bef, NV, FC);
  gemm(bufA, pWf2, bf2, out, NV, FC, NC, false, true);     // FC2 + final ReLU
}
